// KVCacheModel_87668872446435
// MI455X (gfx1250) — matrix operand order repaired and checked by reading, compiled, not run on hardware
//
#include <hip/hip_runtime.h>
#include <hip/hip_bf16.h>
#include <math.h>

// ---------------- problem constants (from reference) ----------------
static constexpr int B   = 2;
static constexpr int S   = 2048;
static constexpr int DIM = 2048;
static constexpr int H   = 16;
static constexpr int HD  = 128;          // DIM / H
static constexpr int M   = B * S;        // 4096 rows of activations
static constexpr int FF  = 4 * DIM;      // 8192

typedef __attribute__((ext_vector_type(16))) _Float16 v16h;
typedef __attribute__((ext_vector_type(8)))  float    v8f;
typedef __attribute__((ext_vector_type(4)))  int      v4i;

// pointer-to-int4 in global (AS1) and LDS (AS3) address spaces
typedef __attribute__((address_space(1))) v4i* gptr4;
typedef __attribute__((address_space(3))) v4i* lptr4;

union Frag { v16h v; _Float16 h[16]; };

// ---------------------------------------------------------------------
// CDNA5 async global->LDS copy (16 bytes per lane) + counter wait.
// Builtin confirmed present; signature: (AS1 int4*, AS3 int4*, off, cpol)
// ---------------------------------------------------------------------
__device__ __forceinline__ void async_copy_b128(const _Float16* gsrc, _Float16* ldst)
{
#if __has_builtin(__builtin_amdgcn_global_load_async_to_lds_b128)
    __builtin_amdgcn_global_load_async_to_lds_b128(
        (gptr4)(unsigned long long)(uintptr_t)gsrc,
        (lptr4)(unsigned)(uintptr_t)ldst,     // generic LDS addr low 32 = LDS offset
        0, 0);
#else
    unsigned lo = (unsigned)(uintptr_t)ldst;
    asm volatile("global_load_async_to_lds_b128 %0, %1, off"
                 :: "v"(lo), "v"(gsrc) : "memory");
#endif
}

__device__ __forceinline__ void async_wait_all()
{
    asm volatile("s_wait_asynccnt 0x0" ::: "memory");
}

// =====================================================================
// fp32 -> fp16 bulk convert (weights), 4 elements/thread
// =====================================================================
__global__ __launch_bounds__(256)
void cvt_f16_kernel(const float* __restrict__ src, _Float16* __restrict__ dst, int n)
{
    const int i = (blockIdx.x * 256 + threadIdx.x) * 4;
    if (i + 3 < n) {
        const float4 v = *(const float4*)(src + i);
        dst[i + 0] = (_Float16)v.x;
        dst[i + 1] = (_Float16)v.y;
        dst[i + 2] = (_Float16)v.z;
        dst[i + 3] = (_Float16)v.w;
    }
}

// =====================================================================
// LayerNorm: one 256-thread block per row, fp32 in, TO out (f16 here)
// =====================================================================
template <typename TO>
__global__ __launch_bounds__(256)
void ln_kernel(const float* __restrict__ x, const float* __restrict__ g,
               const float* __restrict__ be, TO* __restrict__ out)
{
    __shared__ float red0[8];
    __shared__ float red1[8];
    const int row = blockIdx.x;
    const float* xr = x + (size_t)row * DIM;
    float s = 0.f, s2 = 0.f;
    for (int i = threadIdx.x; i < DIM; i += 256) {
        float v = xr[i]; s += v; s2 += v * v;
    }
    #pragma unroll
    for (int off = 16; off >= 1; off >>= 1) {
        s  += __shfl_xor(s,  off, 32);
        s2 += __shfl_xor(s2, off, 32);
    }
    const int wv = threadIdx.x >> 5, lane = threadIdx.x & 31;
    if (lane == 0) { red0[wv] = s; red1[wv] = s2; }
    __syncthreads();
    float ts = 0.f, ts2 = 0.f;
    #pragma unroll
    for (int i = 0; i < 8; i++) { ts += red0[i]; ts2 += red1[i]; }
    const float mu   = ts * (1.0f / DIM);
    const float var  = ts2 * (1.0f / DIM) - mu * mu;
    const float rstd = rsqrtf(var + 1e-5f);
    TO* outr = out + (size_t)row * DIM;
    for (int i = threadIdx.x; i < DIM; i += 256) {
        outr[i] = (TO)((xr[i] - mu) * rstd * g[i] + be[i]);
    }
}

// =====================================================================
// WMMA GEMM: C[M,N] = A[M,K] * W[K,N] + bias (A, W are f16 in global)
//   EPI 1: store f16    EPI 2: gelu -> f16    EPI 3: + residual -> f32
// Block tile 128x128, BK=32, 8 waves in 4(M) x 2(N) grid, 32x64 per wave.
// A tile staged via async b128 copies; W tile staged transposed (sync).
// =====================================================================
static constexpr int BM = 128, BN = 128, BK = 32;

template <int EPI>
__global__ __launch_bounds__(256)
void gemm_wmma(const _Float16* __restrict__ A, const _Float16* __restrict__ W,
               const float* __restrict__ bias, const float* __restrict__ res,
               void* __restrict__ outp, int Mi, int Ni, int Ki)
{
    __shared__ alignas(16) _Float16 sA[BM * BK];   // [m][k], stride 32 halfs
    __shared__ alignas(16) _Float16 sBt[BN * BK];  // [n][k], stride 32 halfs

    const int tid  = threadIdx.x;
    const int lane = tid & 31, wv = tid >> 5;
    const int hi   = (lane >> 4) & 1, ln = lane & 15;
    const int wm   = wv & 3, wn = wv >> 2;            // 4 x 2 waves
    const int gm   = blockIdx.y * BM, gn = blockIdx.x * BN;

    v8f acc[2][4];
    #pragma unroll
    for (int c = 0; c < 4; c++) {
        const float bv = bias[gn + wn * 64 + c * 16 + ln];
        #pragma unroll
        for (int a = 0; a < 2; a++)
            #pragma unroll
            for (int r = 0; r < 8; r++) acc[a][c][r] = bv;
    }

    const int brow = tid >> 3, bseg = tid & 7;   // W fill: 32 rows x 8 x 16 halfs

    for (int k0 = 0; k0 < Ki; k0 += BK) {
        // stage A tile via async b128: 4096 halfs = 512 chunks, 2 per thread
        #pragma unroll
        for (int j = 0; j < 2; j++) {
            const int f0  = (tid * 2 + j) * 8;      // half index in sA
            const int row = f0 >> 5, c = f0 & 31;
            async_copy_b128(A + (size_t)(gm + row) * Ki + k0 + c, sA + f0);
        }
        {   // stage W transposed: sBt[n][k]
            const _Float16* src = W + (size_t)(k0 + brow) * Ni + gn + bseg * 16;
            #pragma unroll
            for (int i = 0; i < 16; i++)
                sBt[(bseg * 16 + i) * BK + brow] = src[i];
            if (k0 + BK < Ki) __builtin_prefetch((const void*)(src + (size_t)BK * Ni), 0, 1);
        }
        async_wait_all();
        __syncthreads();

        Frag af[2], bf[4];
        #pragma unroll
        for (int a = 0; a < 2; a++) {
            const _Float16* p = sA + (wm * 32 + a * 16 + ln) * BK + hi * 8;
            #pragma unroll
            for (int j = 0; j < 8; j++) { af[a].h[j] = p[j]; af[a].h[8 + j] = p[16 + j]; }
        }
        #pragma unroll
        for (int c = 0; c < 4; c++) {
            const _Float16* p = sBt + (wn * 64 + c * 16 + ln) * BK + hi * 8;
            #pragma unroll
            for (int j = 0; j < 8; j++) { bf[c].h[j] = p[j]; bf[c].h[8 + j] = p[16 + j]; }
        }
        #pragma unroll
        for (int a = 0; a < 2; a++)
            #pragma unroll
            for (int c = 0; c < 4; c++)
                acc[a][c] = __builtin_amdgcn_wmma_f32_16x16x32_f16(
                    false, af[a].v, false, bf[c].v, (short)0, acc[a][c], false, false);
        __syncthreads();
    }

    #pragma unroll
    for (int a = 0; a < 2; a++)
        #pragma unroll
        for (int c = 0; c < 4; c++) {
            const int col = gn + wn * 64 + c * 16 + ln;
            #pragma unroll
            for (int r = 0; r < 8; r++) {
                const int row = gm + wm * 32 + a * 16 + r + 8 * hi;
                const size_t idx = (size_t)row * Ni + col;
                const float v = acc[a][c][r];
                if (EPI == 1) {
                    ((_Float16*)outp)[idx] = (_Float16)v;
                } else if (EPI == 2) {
                    const float gl = 0.5f * v * (1.0f + erff(v * 0.70710678118654752f));
                    ((_Float16*)outp)[idx] = (_Float16)gl;
                } else {
                    ((float*)outp)[idx] = v + res[idx];
                }
            }
        }
}

// =====================================================================
// Flash attention (causal): grid (S/128, B*H), 256 threads = 8 waves.
// Each wave owns 16 query rows; 64-key blocks staged in LDS
// (K via async b128 copies, V transposed synchronously).
// q,k,v,o: f16 tensors laid out [B, S, DIM] with head h at cols h*HD.
// =====================================================================
__global__ __launch_bounds__(256)
void attn_kernel(const _Float16* __restrict__ q, const _Float16* __restrict__ k,
                 const _Float16* __restrict__ v, _Float16* __restrict__ o)
{
    __shared__ alignas(16) _Float16 sK [64 * 128];    // [key][hd]
    __shared__ alignas(16) _Float16 sVt[128 * 64];    // [hd][key]
    __shared__ alignas(16) _Float16 sP [8 * 16 * 64]; // per-wave probs [qrow][key]

    const int tid  = threadIdx.x;
    const int lane = tid & 31, wv = tid >> 5;
    const int hi   = (lane >> 4) & 1, ln = lane & 15;
    const int qb   = blockIdx.x * 128;
    const int bh   = blockIdx.y;
    const int b    = bh >> 4, h = bh & 15;
    const size_t base = ((size_t)b * S) * DIM + (size_t)h * HD;
    const float scale = 0.08838834764831845f;  // 1/sqrt(128)

    // preload Q A-fragments (4 k-steps over HD=128)
    Frag qf[4];
    {
        const _Float16* qp = q + base + (size_t)(qb + wv * 16 + ln) * DIM;
        #pragma unroll
        for (int kk = 0; kk < 4; kk++) {
            const _Float16* p = qp + kk * 32 + hi * 8;
            #pragma unroll
            for (int j = 0; j < 8; j++) { qf[kk].h[j] = p[j]; qf[kk].h[8 + j] = p[16 + j]; }
        }
    }

    v8f oacc[8];
    #pragma unroll
    for (int nt = 0; nt < 8; nt++)
        #pragma unroll
        for (int r = 0; r < 8; r++) oacc[nt][r] = 0.f;
    float rmax[8], rsum[8];
    #pragma unroll
    for (int r = 0; r < 8; r++) { rmax[r] = -3.0e38f; rsum[r] = 0.f; }

    const int nkb = qb / 64 + 2;   // causal: only key blocks touching qb..qb+127
    for (int kb = 0; kb < nkb; kb++) {
        const int kg0 = kb * 64;
        // stage K block [64][128] via async b128: 8192 halfs = 1024 chunks, 4/thread
        #pragma unroll
        for (int j = 0; j < 4; j++) {
            const int f0  = (tid * 4 + j) * 8;
            const int row = f0 >> 7, c = f0 & 127;
            async_copy_b128(k + base + (size_t)(kg0 + row) * DIM + c, sK + f0);
        }
        {   // stage V block transposed: sVt[hd][key]
            const int row = tid >> 2, seg = tid & 3;
            const _Float16* src = v + base + (size_t)(kg0 + row) * DIM + seg * 32;
            #pragma unroll
            for (int i = 0; i < 32; i++) sVt[(seg * 32 + i) * 64 + row] = src[i];
            if (kb + 1 < nkb) __builtin_prefetch((const void*)(src + (size_t)64 * DIM), 0, 1);
        }
        async_wait_all();
        __syncthreads();

        // scores: 4 n-tiles x 4 k-steps of WMMA
        v8f sc[4];
        #pragma unroll
        for (int c = 0; c < 4; c++)
            #pragma unroll
            for (int r = 0; r < 8; r++) sc[c][r] = 0.f;
        #pragma unroll
        for (int kk = 0; kk < 4; kk++) {
            #pragma unroll
            for (int c = 0; c < 4; c++) {
                Frag bfr;
                const _Float16* p = sK + (c * 16 + ln) * 128 + kk * 32 + hi * 8;
                #pragma unroll
                for (int j = 0; j < 8; j++) { bfr.h[j] = p[j]; bfr.h[8 + j] = p[16 + j]; }
                sc[c] = __builtin_amdgcn_wmma_f32_16x16x32_f16(
                    false, qf[kk].v, false, bfr.v, (short)0, sc[c], false, false);
            }
        }

        // causal mask + online softmax (rows live in 16-lane groups)
        float corr[8];
        #pragma unroll
        for (int r = 0; r < 8; r++) {
            const int lr = r + 8 * hi;               // local row in wave tile
            const int qg = qb + wv * 16 + lr;        // global query index
            float vals[4];
            float cm = -3.0e38f;
            #pragma unroll
            for (int c = 0; c < 4; c++) {
                const int kg = kg0 + c * 16 + ln;
                float sv = sc[c][r] * scale;
                if (kg > qg) sv = -1.0e9f;
                vals[c] = sv;
                cm = fmaxf(cm, sv);
            }
            #pragma unroll
            for (int off = 8; off >= 1; off >>= 1) cm = fmaxf(cm, __shfl_xor(cm, off, 16));
            const float mnew = fmaxf(rmax[r], cm);
            const float cr   = __expf(rmax[r] - mnew);
            rmax[r] = mnew; corr[r] = cr;
            float ps = 0.f;
            #pragma unroll
            for (int c = 0; c < 4; c++) {
                const float pv = __expf(vals[c] - mnew);
                ps += pv;
                sP[wv * 1024 + lr * 64 + c * 16 + ln] = (_Float16)pv;
            }
            #pragma unroll
            for (int off = 8; off >= 1; off >>= 1) ps += __shfl_xor(ps, off, 16);
            rsum[r] = rsum[r] * cr + ps;
        }
        asm volatile("s_wait_dscnt 0x0" ::: "memory");  // wave's P writes -> P reads

        // rescale running O
        #pragma unroll
        for (int nt = 0; nt < 8; nt++)
            #pragma unroll
            for (int r = 0; r < 8; r++) oacc[nt][r] *= corr[r];

        // O += P x V  (2 key k-steps x 8 hd tiles)
        Frag pf[2];
        #pragma unroll
        for (int kk2 = 0; kk2 < 2; kk2++) {
            const _Float16* p = sP + wv * 1024 + ln * 64 + kk2 * 32 + hi * 8;
            #pragma unroll
            for (int j = 0; j < 8; j++) { pf[kk2].h[j] = p[j]; pf[kk2].h[8 + j] = p[16 + j]; }
        }
        #pragma unroll
        for (int nt = 0; nt < 8; nt++) {
            #pragma unroll
            for (int kk2 = 0; kk2 < 2; kk2++) {
                Frag bfr;
                const _Float16* p2 = sVt + (nt * 16 + ln) * 64 + kk2 * 32 + hi * 8;
                #pragma unroll
                for (int j = 0; j < 8; j++) { bfr.h[j] = p2[j]; bfr.h[8 + j] = p2[16 + j]; }
                oacc[nt] = __builtin_amdgcn_wmma_f32_16x16x32_f16(
                    false, pf[kk2].v, false, bfr.v, (short)0, oacc[nt], false, false);
            }
        }
        __syncthreads();
    }

    // write O / l  -> attn buffer (f16, [B,S,DIM] head-major cols)
    #pragma unroll
    for (int nt = 0; nt < 8; nt++)
        #pragma unroll
        for (int r = 0; r < 8; r++) {
            const int row = qb + wv * 16 + r + 8 * hi;
            const float ov = oacc[nt][r] / rsum[r];
            o[base + (size_t)row * DIM + nt * 16 + ln] = (_Float16)ov;
        }
}

// =====================================================================
// launch
// =====================================================================
extern "C" void kernel_launch(void* const* d_in, const int* in_sizes, int n_in,
                              void* d_out, int out_size, void* d_ws, size_t ws_size,
                              hipStream_t stream)
{
    (void)in_sizes; (void)n_in; (void)out_size; (void)ws_size;
    const float* x   = (const float*)d_in[0];
    const float* wq  = (const float*)d_in[1];
    const float* bq  = (const float*)d_in[2];
    const float* wk  = (const float*)d_in[3];
    const float* bk  = (const float*)d_in[4];
    const float* wvp = (const float*)d_in[5];
    const float* bv  = (const float*)d_in[6];
    const float* wo  = (const float*)d_in[7];
    const float* bo  = (const float*)d_in[8];
    const float* w1  = (const float*)d_in[9];
    const float* b1  = (const float*)d_in[10];
    const float* w2  = (const float*)d_in[11];
    const float* b2  = (const float*)d_in[12];
    const float* g1  = (const float*)d_in[13];
    const float* be1 = (const float*)d_in[14];
    const float* g2  = (const float*)d_in[15];
    const float* be2 = (const float*)d_in[16];
    float* out = (float*)d_out;

    // workspace layout (all f16 except residual stream)
    char* w = (char*)d_ws;
    size_t off = 0;
    _Float16* wqh  = (_Float16*)(w + off); off += (size_t)DIM * DIM * 2;
    _Float16* wkh  = (_Float16*)(w + off); off += (size_t)DIM * DIM * 2;
    _Float16* wvh  = (_Float16*)(w + off); off += (size_t)DIM * DIM * 2;
    _Float16* woh  = (_Float16*)(w + off); off += (size_t)DIM * DIM * 2;
    _Float16* w1h  = (_Float16*)(w + off); off += (size_t)DIM * FF  * 2;
    _Float16* w2h  = (_Float16*)(w + off); off += (size_t)FF  * DIM * 2;
    _Float16* nx   = (_Float16*)(w + off); off += (size_t)M * DIM * 2;
    _Float16* qh   = (_Float16*)(w + off); off += (size_t)M * DIM * 2;
    _Float16* kh   = (_Float16*)(w + off); off += (size_t)M * DIM * 2;
    _Float16* vh   = (_Float16*)(w + off); off += (size_t)M * DIM * 2;
    _Float16* atth = (_Float16*)(w + off); off += (size_t)M * DIM * 2;
    float*    xres = (float*)(w + off);    off += (size_t)M * DIM * 4;
    _Float16* nx2  = (_Float16*)(w + off); off += (size_t)M * DIM * 2;
    _Float16* hbuf = (_Float16*)(w + off); off += (size_t)M * FF  * 2;

    const dim3 blk(256);

    // 0. weights fp32 -> fp16 (one pass)
    const int nW = DIM * DIM, nF = DIM * FF;
    cvt_f16_kernel<<<dim3(nW / 1024), blk, 0, stream>>>(wq,  wqh, nW);
    cvt_f16_kernel<<<dim3(nW / 1024), blk, 0, stream>>>(wk,  wkh, nW);
    cvt_f16_kernel<<<dim3(nW / 1024), blk, 0, stream>>>(wvp, wvh, nW);
    cvt_f16_kernel<<<dim3(nW / 1024), blk, 0, stream>>>(wo,  woh, nW);
    cvt_f16_kernel<<<dim3(nF / 1024), blk, 0, stream>>>(w1,  w1h, nF);
    cvt_f16_kernel<<<dim3(nF / 1024), blk, 0, stream>>>(w2,  w2h, nF);

    // 1. LN1 -> f16
    ln_kernel<_Float16><<<dim3(M), blk, 0, stream>>>(x, g1, be1, nx);

    // 2. Q, K, V projections (f16 out for attention)
    const dim3 gP(DIM / BN, M / BM);
    gemm_wmma<1><<<gP, blk, 0, stream>>>(nx, wqh, bq, nullptr, (void*)qh, M, DIM, DIM);
    gemm_wmma<1><<<gP, blk, 0, stream>>>(nx, wkh, bk, nullptr, (void*)kh, M, DIM, DIM);
    gemm_wmma<1><<<gP, blk, 0, stream>>>(nx, wvh, bv, nullptr, (void*)vh, M, DIM, DIM);

    // 3. causal flash attention
    attn_kernel<<<dim3(S / 128, B * H), blk, 0, stream>>>(qh, kh, vh, atth);

    // 4. output projection + residual -> xres (f32)
    gemm_wmma<3><<<gP, blk, 0, stream>>>(atth, woh, bo, x, (void*)xres, M, DIM, DIM);

    // 5. LN2 -> f16
    ln_kernel<_Float16><<<dim3(M), blk, 0, stream>>>(xres, g2, be2, nx2);

    // 6. FFN up + exact GELU -> f16
    gemm_wmma<2><<<dim3(FF / BN, M / BM), blk, 0, stream>>>(
        nx2, w1h, b1, nullptr, (void*)hbuf, M, FF, DIM);

    // 7. FFN down + residual -> d_out (f32)
    gemm_wmma<3><<<gP, blk, 0, stream>>>(hbuf, w2h, b2, xres, (void*)out, M, DIM, FF);
}
